// PerNodeMemory_84275848282474
// MI455X (gfx1250) — matrix operand order repaired and checked by reading, not compiled, not run on hardware
//
#include <hip/hip_runtime.h>
#include <hip/hip_bf16.h>

// ---------- vector types (plain ext_vector to keep unions trivial) ----------
typedef __attribute__((ext_vector_type(16))) __bf16        v16bf;
typedef __attribute__((ext_vector_type(8)))  float         v8f;
typedef __attribute__((ext_vector_type(4)))  unsigned int  u32x4;
typedef __attribute__((ext_vector_type(2)))  unsigned int  u32x2;

union ABfrag { v16bf v; u32x4 q[2]; };

__device__ __forceinline__ unsigned short f2bf(float f) {
    unsigned int u = __float_as_uint(f);
    unsigned int r = u + 0x7FFFu + ((u >> 16) & 1u);   // RNE
    return (unsigned short)(r >> 16);
}

// ---------- prep kernel 1: f32 rows -> bf16 rows + sum of squares ----------
// one wave per 128-element row; lane handles 4 elements
__global__ void __launch_bounds__(256) row_convert_sumsq(
    const float* __restrict__ src, unsigned short* __restrict__ dst,
    float* __restrict__ sums, int rows)
{
    const int lane = threadIdx.x & 31;
    const int wave = threadIdx.x >> 5;
    const int row  = blockIdx.x * 8 + wave;
    if (row >= rows) return;

    const float4 v = ((const float4*)(src + (size_t)row * 128))[lane];
    float ss = v.x * v.x + v.y * v.y + v.z * v.z + v.w * v.w;

    u32x2 pk;
    pk.x = (unsigned)f2bf(v.x) | ((unsigned)f2bf(v.y) << 16);
    pk.y = (unsigned)f2bf(v.z) | ((unsigned)f2bf(v.w) << 16);
    ((u32x2*)(dst + (size_t)row * 128))[lane] = pk;

    ss += __shfl_xor(ss, 1, 32);
    ss += __shfl_xor(ss, 2, 32);
    ss += __shfl_xor(ss, 4, 32);
    ss += __shfl_xor(ss, 8, 32);
    ss += __shfl_xor(ss, 16, 32);
    if (lane == 0) sums[row] = ss;
}

// ---------- prep kernel 2: data[S][128] f32 -> dataT[128][S] bf16 ----------
__global__ void __launch_bounds__(256) transpose_bf(
    const float* __restrict__ data, unsigned short* __restrict__ dataT, int S)
{
    __shared__ unsigned short tile[32][33];
    const int tx = threadIdx.x & 31;
    const int ty = threadIdx.x >> 5;              // 0..7
    const int sTiles = S >> 5;
    const int s0 = (blockIdx.x % sTiles) * 32;
    const int d0 = (blockIdx.x / sTiles) * 32;

#pragma unroll
    for (int i = 0; i < 4; ++i) {
        int s = ty + i * 8;
        tile[s][tx] = f2bf(data[(size_t)(s0 + s) * 128 + d0 + tx]);
    }
    __syncthreads();
#pragma unroll
    for (int i = 0; i < 4; ++i) {
        int d = ty + i * 8;
        dataT[(size_t)(d0 + d) * S + s0 + tx] = tile[tx][d];
    }
}

// ---------- main kernel: flash-style softmax-weighted memory readout ----------
// 1 wave handles 16 query rows over all S; 8 waves per block.
__global__ void __launch_bounds__(256) pernode_main(
    const float* __restrict__ node,            // f32 residual
    const unsigned short* __restrict__ node_bf,// [BN][128] bf16
    const unsigned short* __restrict__ data_bf,// [S][128] bf16
    const unsigned short* __restrict__ dataT,  // [128][S] bf16
    const float* __restrict__ q2,              // [BN]
    const float* __restrict__ m2,              // [S]
    const float* __restrict__ tempP,
    const float* __restrict__ lerpP,
    float* __restrict__ out, int S, int BN)
{
    const int lane    = threadIdx.x & 31;
    const int wave    = threadIdx.x >> 5;
    const int rowBase = (blockIdx.x * 8 + wave) * 16;
    if (rowBase + 16 > BN) return;              // wave-uniform
    const int n16  = lane & 15;
    const int half = lane >> 4;

    __shared__ __align__(16) unsigned short Ptile[8][16 * 32];
    unsigned short* P = Ptile[wave];

    const float temp = tempP[0];

    // Resident A operands: Q tile 16x128 as 4 chunks of 16x32 bf16.
    // A layout: lane m=n16; half 0 -> K kc*32+0..15, half 1 -> K kc*32+16..31.
    ABfrag aq[4];
    {
        const unsigned short* qrow = node_bf + (size_t)(rowBase + n16) * 128;
#pragma unroll
        for (int kc = 0; kc < 4; ++kc) {
            const int c1 = kc * 32 + half * 16;
            aq[kc].q[0] = *(const u32x4*)(qrow + c1);
            aq[kc].q[1] = *(const u32x4*)(qrow + c1 + 8);
        }
    }

    float q2v[8];
#pragma unroll
    for (int r = 0; r < 8; ++r) q2v[r] = q2[rowBase + r + 8 * half];

    v8f accs[8];
#pragma unroll
    for (int t = 0; t < 8; ++t) accs[t] = v8f{0.f,0.f,0.f,0.f,0.f,0.f,0.f,0.f};
    float rowsum[8] = {0.f,0.f,0.f,0.f,0.f,0.f,0.f,0.f};

    for (int s0 = 0; s0 < S; s0 += 32) {
        // ---- GEMM1: cross[16 x 32] = Q(16x128) . data(s0..s0+31, :)^T ----
        v8f c0 = v8f{0.f,0.f,0.f,0.f,0.f,0.f,0.f,0.f};
        v8f c1 = v8f{0.f,0.f,0.f,0.f,0.f,0.f,0.f,0.f};
        const unsigned short* brow0 = data_bf + (size_t)(s0 + n16) * 128;
        const unsigned short* brow1 = data_bf + (size_t)(s0 + 16 + n16) * 128;
#pragma unroll
        for (int kc = 0; kc < 4; ++kc) {
            // B layout: lane n=n16 holds K = half*16 .. half*16+15 contiguous
            const int cb = kc * 32 + half * 16;
            ABfrag b0, b1;
            b0.q[0] = *(const u32x4*)(brow0 + cb);
            b0.q[1] = *(const u32x4*)(brow0 + cb + 8);
            b1.q[0] = *(const u32x4*)(brow1 + cb);
            b1.q[1] = *(const u32x4*)(brow1 + cb + 8);
            c0 = __builtin_amdgcn_wmma_f32_16x16x32_bf16(
                     false, aq[kc].v, false, b0.v, (short)0, c0, false, false);
            c1 = __builtin_amdgcn_wmma_f32_16x16x32_bf16(
                     false, aq[kc].v, false, b1.v, (short)0, c1, false, false);
        }

        // ---- elementwise: d2 -> ds -> s=exp(-t*ds) -> p=exp(s) ----
        const float m2v0 = m2[s0 + n16];
        const float m2v1 = m2[s0 + 16 + n16];
#pragma unroll
        for (int r = 0; r < 8; ++r) {
            const int prow = (r + 8 * half) * 32;
            float d2a = q2v[r] + m2v0 - 2.0f * c0[r];
            float dsa = __builtin_amdgcn_sqrtf(fmaxf(d2a, 1e-12f));
            float pa  = __expf(__expf(-temp * dsa));
            rowsum[r] += pa;
            P[prow + n16] = f2bf(pa);

            float d2b = q2v[r] + m2v1 - 2.0f * c1[r];
            float dsb = __builtin_amdgcn_sqrtf(fmaxf(d2b, 1e-12f));
            float pb  = __expf(__expf(-temp * dsb));
            rowsum[r] += pb;
            P[prow + 16 + n16] = f2bf(pb);
        }

        // wave-private LDS round trip to re-stripe P into A-operand layout
        asm volatile("s_wait_dscnt 0" ::: "memory");
        ABfrag pa;
        pa.q[0] = *(const u32x4*)(P + n16 * 32 + half * 16);
        pa.q[1] = *(const u32x4*)(P + n16 * 32 + half * 16 + 8);

        // ---- GEMM2: acc[16 x 128] += P(16x32) . data(s0..s0+31, :) ----
#pragma unroll
        for (int t = 0; t < 8; ++t) {
            const unsigned short* trow =
                dataT + (size_t)(t * 16 + n16) * S + s0 + half * 16;
            ABfrag bd;
            bd.q[0] = *(const u32x4*)(trow);
            bd.q[1] = *(const u32x4*)(trow + 8);
            accs[t] = __builtin_amdgcn_wmma_f32_16x16x32_bf16(
                          false, pa.v, false, bd.v, (short)0, accs[t], false, false);
        }
    }

    // ---- softmax denominator: reduce across the 16-lane half sharing row m ----
#pragma unroll
    for (int r = 0; r < 8; ++r) {
        float v = rowsum[r];
        v += __shfl_xor(v, 1, 32);
        v += __shfl_xor(v, 2, 32);
        v += __shfl_xor(v, 4, 32);
        v += __shfl_xor(v, 8, 32);
        rowsum[r] = 1.0f / v;
    }

    const float fl   = lerpP[0];
    const float lerp = 1.0f / (1.0f + __expf(-fl));
    const float om   = 1.0f - lerp;

#pragma unroll
    for (int t = 0; t < 8; ++t) {
#pragma unroll
        for (int r = 0; r < 8; ++r) {
            const int m = r + 8 * half;
            const int d = t * 16 + n16;
            const size_t idx = (size_t)(rowBase + m) * 128 + d;
            out[idx] = lerp * accs[t][r] * rowsum[r] + om * node[idx];
        }
    }
}

extern "C" void kernel_launch(void* const* d_in, const int* in_sizes, int n_in,
                              void* d_out, int out_size, void* d_ws, size_t ws_size,
                              hipStream_t stream) {
    const float* node  = (const float*)d_in[0];
    const float* data  = (const float*)d_in[1];
    const float* temp  = (const float*)d_in[2];
    const float* flerp = (const float*)d_in[3];
    float* out = (float*)d_out;

    const int BN = in_sizes[0] / 128;   // 16384
    const int S  = in_sizes[1] / 128;   // 4096

    char* ws = (char*)d_ws;
    unsigned short* node_bf = (unsigned short*)ws;
    size_t off = (size_t)BN * 128 * 2;
    float* q2 = (float*)(ws + off);           off += (size_t)BN * 4;
    unsigned short* data_bf = (unsigned short*)(ws + off); off += (size_t)S * 128 * 2;
    float* m2 = (float*)(ws + off);           off += (size_t)S * 4;
    unsigned short* dataT = (unsigned short*)(ws + off);

    row_convert_sumsq<<<(BN + 7) / 8, 256, 0, stream>>>(node, node_bf, q2, BN);
    row_convert_sumsq<<<(S + 7) / 8, 256, 0, stream>>>(data, data_bf, m2, S);
    transpose_bf<<<(S / 32) * 4, 256, 0, stream>>>(data, dataT, S);

    pernode_main<<<BN / 128, 256, 0, stream>>>(
        node, node_bf, data_bf, dataT, q2, m2, temp, flerp, out, S, BN);
}
